// RotateLinearInt4_601295421924
// MI455X (gfx1250) — matrix operand order repaired and checked by reading, not compiled, not run on hardware
//
#include <hip/hip_runtime.h>
#include <hip/hip_bf16.h>

#define TOKENS   8192
#define IN_FEAT  4096
#define OUT_FEAT 4096
#define GROUP    128
#define KROT     8
#define NPAIR    (IN_FEAT / 2)     // 2048
#define NGROUPS  (IN_FEAT / GROUP) // 32

typedef __bf16 v8bf  __attribute__((ext_vector_type(8)));
typedef __bf16 v16bf __attribute__((ext_vector_type(16)));
typedef float  v8f   __attribute__((ext_vector_type(8)));
typedef int    v4i   __attribute__((vector_size(16)));   // matches builtin pointee

// AS1/AS3 pointer materialization (inttoptr: generic LDS ptr low 32 bits = LDS offset)
#define AS1P(p) ((__attribute__((address_space(1))) v4i*)(unsigned long long)(p))
#define AS3P(p) ((__attribute__((address_space(3))) v4i*)(unsigned long long)(p))

#if __has_builtin(__builtin_amdgcn_global_load_async_to_lds_b128) && \
    __has_builtin(__builtin_amdgcn_s_wait_asynccnt)
#define HAVE_ASYNC_LDS 1
#else
#define HAVE_ASYNC_LDS 0
#endif

// ---------------------------------------------------------------------------
// Kernel 1: precompute cos/sin tables for the 8 rotation passes (8 x 2048)
// ---------------------------------------------------------------------------
__global__ __launch_bounds__(256) void cs_kernel(const float* __restrict__ theta,
                                                 float* __restrict__ cosT,
                                                 float* __restrict__ sinT) {
    int i = blockIdx.x * 256 + threadIdx.x;
    if (i < KROT * NPAIR) {
        float s, c;
        __sincosf(theta[i], &s, &c);
        cosT[i] = c;
        sinT[i] = s;
    }
}

// ---------------------------------------------------------------------------
// Kernel 2: per-token Givens rotations in LDS, then channel scale -> bf16
// ---------------------------------------------------------------------------
__global__ __launch_bounds__(256) void rotate_kernel(const float* __restrict__ x,
                                                     const int*   __restrict__ pairs,
                                                     const float* __restrict__ cosT,
                                                     const float* __restrict__ sinT,
                                                     const float* __restrict__ chs,
                                                     __bf16*      __restrict__ xr) {
    __shared__ float row[IN_FEAT];                 // 16 KB
    const int token = blockIdx.x;
    const int tid   = threadIdx.x;

    const float4* xin  = (const float4*)(x + (size_t)token * IN_FEAT);
    float4*       rowv = (float4*)row;
    for (int s = 0; s < 4; ++s) rowv[s * 256 + tid] = xin[s * 256 + tid];
    __syncthreads();

    for (int k = 0; k < KROT; ++k) {
        const int*   pr = pairs + k * IN_FEAT;
        const float* cT = cosT + k * NPAIR;
        const float* sT = sinT + k * NPAIR;
        for (int pp = 0; pp < NPAIR / 256; ++pp) {
            int   p   = pp * 256 + tid;
            int   off = (p >> 6) << 7;             // (pair_idx/64)*128 group offset
            int   i   = pr[2 * p]     + off;
            int   j   = pr[2 * p + 1] + off;
            float c   = cT[p], s = sT[p];
            float xi  = row[i], xj = row[j];
            row[i] = c * xi - s * xj;
            row[j] = s * xi + c * xj;
        }
        __syncthreads();
    }

    __bf16* outp = xr + (size_t)token * IN_FEAT;
    int base = tid * 16;
    v8bf o0, o1;
    for (int e = 0; e < 8; ++e) o0[e] = (__bf16)(row[base + e]     * chs[base + e]);
    for (int e = 0; e < 8; ++e) o1[e] = (__bf16)(row[base + 8 + e] * chs[base + 8 + e]);
    *(v8bf*)(outp + base)     = o0;
    *(v8bf*)(outp + base + 8) = o1;
}

// ---------------------------------------------------------------------------
// Kernel 3: dequantize int4-in-int32 weights to bf16  (N x K, K contiguous)
// ---------------------------------------------------------------------------
__global__ __launch_bounds__(256) void dequant_kernel(const int*   __restrict__ q,
                                                      const float* __restrict__ scales,
                                                      const float* __restrict__ zeros,
                                                      __bf16*      __restrict__ wb) {
    size_t t    = (size_t)blockIdx.x * 256 + threadIdx.x;
    size_t base = t * 8;
    int    n    = (int)(base / IN_FEAT);
    int    k    = (int)(base % IN_FEAT);
    int    g    = k / GROUP;
    float  sc   = scales[n * NGROUPS + g];
    float  zp   = zeros [n * NGROUPS + g];
    int4 q0 = *(const int4*)(q + base);
    int4 q1 = *(const int4*)(q + base + 4);
    v8bf o;
    o[0] = (__bf16)(((float)q0.x - zp) * sc);
    o[1] = (__bf16)(((float)q0.y - zp) * sc);
    o[2] = (__bf16)(((float)q0.z - zp) * sc);
    o[3] = (__bf16)(((float)q0.w - zp) * sc);
    o[4] = (__bf16)(((float)q1.x - zp) * sc);
    o[5] = (__bf16)(((float)q1.y - zp) * sc);
    o[6] = (__bf16)(((float)q1.z - zp) * sc);
    o[7] = (__bf16)(((float)q1.w - zp) * sc);
    *(v8bf*)(wb + base) = o;
}

// ---------------------------------------------------------------------------
// Kernel 4: C[M,N] = A[M,K](bf16) * B[N,K](bf16)^T, v_wmma_f32_16x16x32_bf16.
// Block: 256 threads = 8 wave32, tile 64(M) x 256(N), K-step 32.
// Wave grid 2(M) x 4(N); each wave owns a 32x64 patch = 2x4 WMMA tiles.
// Double-buffered LDS; staging via GLOBAL_LOAD_ASYNC_TO_LDS_B128 (ASYNCcnt)
// when the toolchain declares the builtin, else global->VGPR->ds_store.
// ---------------------------------------------------------------------------
#define BM 64
#define BN 256
#define BK 32
#define LDK (BK + 8)   // row stride 80 bytes -> all 16B-aligned accesses
#define NKSTEPS (IN_FEAT / BK)

__device__ __forceinline__ void copy16_async(const __bf16* g, __bf16* l) {
#if HAVE_ASYNC_LDS
    __builtin_amdgcn_global_load_async_to_lds_b128(AS1P(g), AS3P(l), 0, 0);
#else
    *(v8bf*)l = *(const v8bf*)g;
#endif
}

__global__ __launch_bounds__(256) void gemm_kernel(const __bf16* __restrict__ A,
                                                   const __bf16* __restrict__ B,
                                                   float*        __restrict__ C) {
    __shared__ __align__(16) __bf16 As[2][BM][LDK];   // 2 x 5  KB
    __shared__ __align__(16) __bf16 Bs[2][BN][LDK];   // 2 x 20 KB

    const int tid  = threadIdx.x;
    const int lane = tid & 31;
    const int wave = tid >> 5;
    const int wm   = wave & 1;        // M sub-block: 0..1 (x32)
    const int wn   = wave >> 1;       // N sub-block: 0..3 (x64)
    const int m0   = blockIdx.y * BM;
    const int n0   = blockIdx.x * BN;

    v8f acc[2][4];
    for (int mi = 0; mi < 2; ++mi)
        for (int ni = 0; ni < 4; ++ni)
            for (int e = 0; e < 8; ++e) acc[mi][ni][e] = 0.0f;

    // staging map: A -> 1 chunk/thread (64 rows x 4 chunks); B -> 4 chunks/thread
    const int arow = tid >> 2;
    const int acol = (tid & 3) * 8;

    // fragment addressing (ISA VGPR layouts, wave32):
    //  A 16x32: lane<16 -> K chunks {0..7,16..23}; lane>=16 -> {8..15,24..31}
    //  B 32x16: lane<16 -> N=lane, K {0..7,16..23}; lane>=16 -> N=lane-16, K {8..15,24..31}
    const int mrow = lane & 15;
    const int aklo = (lane < 16) ? 0 : 8;
    const int bk   = (lane < 16) ? 0 : 8;

    // ---- prologue: stage K-tile 0 into buffer 0 ----
    {
        const __bf16* Ag = A + (size_t)(m0 + arow) * IN_FEAT + acol;
        copy16_async(Ag, &As[0][arow][acol]);
        for (int s = 0; s < 4; ++s) {
            int chunk = tid + 256 * s;
            int brow  = chunk >> 2;
            int bcol  = (chunk & 3) * 8;
            copy16_async(B + (size_t)(n0 + brow) * IN_FEAT + bcol, &Bs[0][brow][bcol]);
        }
    }
#if HAVE_ASYNC_LDS
    __builtin_amdgcn_s_wait_asynccnt(0);
#endif
    __syncthreads();

    int cur = 0;
    for (int kt = 0; kt < NKSTEPS; ++kt) {
        const int nxt = cur ^ 1;

        // ---- stage next K-tile into the other buffer (overlaps WMMA below) ----
        if (kt + 1 < NKSTEPS) {
            const int k0n = (kt + 1) * BK;
            copy16_async(A + (size_t)(m0 + arow) * IN_FEAT + k0n + acol,
                         &As[nxt][arow][acol]);
            for (int s = 0; s < 4; ++s) {
                int chunk = tid + 256 * s;
                int brow  = chunk >> 2;
                int bcol  = (chunk & 3) * 8;
                copy16_async(B + (size_t)(n0 + brow) * IN_FEAT + k0n + bcol,
                             &Bs[nxt][brow][bcol]);
            }
        }

        // ---- compute on current buffer ----
        v16bf afrag[2], bfrag[4];
        for (int mi = 0; mi < 2; ++mi) {
            const __bf16* ap = &As[cur][wm * 32 + mi * 16 + mrow][0];
            v8bf lo = *(const v8bf*)(ap + aklo);
            v8bf hi = *(const v8bf*)(ap + aklo + 16);
            afrag[mi] = __builtin_shufflevector(lo, hi, 0, 1, 2, 3, 4, 5, 6, 7,
                                                8, 9, 10, 11, 12, 13, 14, 15);
        }
        for (int ni = 0; ni < 4; ++ni) {
            const __bf16* bp = &Bs[cur][wn * 64 + ni * 16 + mrow][bk];
            v8bf lo = *(const v8bf*)(bp);
            v8bf hi = *(const v8bf*)(bp + 16);
            bfrag[ni] = __builtin_shufflevector(lo, hi, 0, 1, 2, 3, 4, 5, 6, 7,
                                                8, 9, 10, 11, 12, 13, 14, 15);
        }
        for (int mi = 0; mi < 2; ++mi)
            for (int ni = 0; ni < 4; ++ni)
                acc[mi][ni] = __builtin_amdgcn_wmma_f32_16x16x32_bf16(
                    /*neg_a=*/false, afrag[mi],
                    /*neg_b=*/false, bfrag[ni],
                    /*c_mod=*/(short)0, acc[mi][ni],
                    /*reuse_a=*/false, /*reuse_b=*/false);

        // ---- retire this wave's async copies, then block-wide handoff ----
#if HAVE_ASYNC_LDS
        __builtin_amdgcn_s_wait_asynccnt(0);
#endif
        __syncthreads();
        cur = nxt;
    }

    // store: C/D layout — VGPR r: lanes 0-15 -> M=r, lanes 16-31 -> M=r+8; N=lane%16
    for (int mi = 0; mi < 2; ++mi)
        for (int ni = 0; ni < 4; ++ni) {
            int rbase = m0 + wm * 32 + mi * 16 + ((lane < 16) ? 0 : 8);
            int col   = n0 + wn * 64 + ni * 16 + (lane & 15);
            for (int r = 0; r < 8; ++r)
                C[(size_t)(rbase + r) * OUT_FEAT + col] = acc[mi][ni][r];
        }
}

// ---------------------------------------------------------------------------
// Host launcher
// ---------------------------------------------------------------------------
extern "C" void kernel_launch(void* const* d_in, const int* in_sizes, int n_in,
                              void* d_out, int out_size, void* d_ws, size_t ws_size,
                              hipStream_t stream) {
    const float* x      = (const float*)d_in[0];
    const float* theta  = (const float*)d_in[1];
    const float* chs    = (const float*)d_in[2];
    const float* scales = (const float*)d_in[3];
    const float* zeros  = (const float*)d_in[4];
    const int*   pairs  = (const int*)d_in[5];
    const int*   qw     = (const int*)d_in[6];
    float*       out    = (float*)d_out;

    // workspace layout: xr(bf16) | wb(bf16) | cosT | sinT  (~96.2 MB total)
    char*   ws   = (char*)d_ws;
    __bf16* xr   = (__bf16*)ws;                                   // 64 MB
    __bf16* wb   = (__bf16*)(ws + (size_t)TOKENS * IN_FEAT * 2);  // 32 MB
    float*  cosT = (float*)(ws + (size_t)TOKENS * IN_FEAT * 2
                               + (size_t)OUT_FEAT * IN_FEAT * 2);
    float*  sinT = cosT + KROT * NPAIR;

    cs_kernel<<<(KROT * NPAIR + 255) / 256, 256, 0, stream>>>(theta, cosT, sinT);

    dequant_kernel<<<(size_t)OUT_FEAT * IN_FEAT / 8 / 256, 256, 0, stream>>>(
        qw, scales, zeros, wb);

    rotate_kernel<<<TOKENS, 256, 0, stream>>>(x, pairs, cosT, sinT, chs, xr);

    dim3 grid(OUT_FEAT / BN, TOKENS / BM);
    gemm_kernel<<<grid, 256, 0, stream>>>(xr, wb, out);
}
